// MultiHeadFast_49263274885988
// MI455X (gfx1250) — compile-verified, matrix operand order checked by reading, not run on hardware
//
#include <hip/hip_runtime.h>
#include <hip/hip_bf16.h>
#include <stdint.h>

// ---------------------------------------------------------------------------
// MultiHeadFast on MI455X (gfx1250, wave32, WMMA bf16 -> f32 accumulate)
//
//   k0  cvt_x     : x fp32 -> bf16
//   k1  cvt_w     : Ws fp32 (1024x3072) -> W^T bf16 (3072x1024) (K-major B)
//   k2  qkv_gemm  : 32x64 wave tile, unroll-by-2 ping-pong fragment sets,
//                   scatter Q,K (B,H,T,64) and V transposed (B,H,64,T)
//   k3  col_stats : per-column softmax stats (softmax over the QUERY axis),
//                   unroll-by-2 Q tiles, ds_swizzle column reduction
//   k4  attn_out  : recompute S, p = exp(s-m[k])*invN[k], LDS transpose of P
//                   into A-fragments, O += P@V ; K frags ping-pong buffered
// ---------------------------------------------------------------------------

#define B_SZ   2
#define T_SZ   2048
#define H_CNT  16
#define D_SZ   64
#define EMB    1024
#define N3     3072            // 3*H*D
#define M_TOK  (B_SZ * T_SZ)   // 4096
#define NQT    (T_SZ / 16)     // 128 query tiles per (b,h)
#define SCALE  0.022097086912079612f   // 1/sqrt(2048)

typedef __attribute__((ext_vector_type(16))) __bf16 v16bf;
typedef __attribute__((ext_vector_type(8)))  __bf16 v8bf;
typedef __attribute__((ext_vector_type(8)))  float  v8f;

// ---- WMMA: D = A(16x32 bf16) * B(32x16 bf16) + C(16x16 f32) ---------------
__device__ __forceinline__ v8f wmma_bf16(v16bf a, v16bf b, v8f c) {
  return __builtin_amdgcn_wmma_f32_16x16x32_bf16(
      false, a, false, b, (short)0, c, false, false);
}

// ---- 16x32 K-major fragment (A layout; B operand = A layout applied to B^T)
__device__ __forceinline__ v16bf load_frag(const __bf16* base, int stride, int lane) {
  const int row = lane & 15;
  const int off = (lane < 16) ? 0 : 8;
  const __bf16* p = base + (size_t)row * stride + off;
  v8bf lo = *(const v8bf*)(p);
  v8bf hi = *(const v8bf*)(p + 16);
  v16bf r;
#pragma unroll
  for (int i = 0; i < 8; ++i) { r[i] = lo[i]; r[i + 8] = hi[i]; }
  return r;
}

// ---- lane l <-> l^16 exchange via ds_swizzle SWAPX16 ----------------------
__device__ __forceinline__ float xor16(float x) {
  int i = __builtin_amdgcn_ds_swizzle(__float_as_int(x), 0x401F);
  return __int_as_float(i);
}

// ===========================================================================
__global__ void cvt_x_kernel(const float* __restrict__ x,
                             __bf16* __restrict__ xb, int n) {
  int i = blockIdx.x * blockDim.x + threadIdx.x;
  if (i < n) xb[i] = (__bf16)x[i];
}

__global__ void cvt_w_kernel(const float* __restrict__ Ws,
                             __bf16* __restrict__ Wt) {
  int i = blockIdx.x * blockDim.x + threadIdx.x;   // index into Wt (3072x1024)
  if (i < N3 * EMB) {
    int nrow = i >> 10;
    int k    = i & 1023;
    Wt[i] = (__bf16)Ws[(size_t)k * N3 + nrow];
  }
}

// ===========================================================================
// QKV GEMM: 32(M) x 64(N) wave tile, unroll-by-2 ping-pong (no rotate movs).
// ===========================================================================
__global__ void __launch_bounds__(32)
qkv_gemm_kernel(const __bf16* __restrict__ xb, const __bf16* __restrict__ Wt,
                __bf16* __restrict__ Qb, __bf16* __restrict__ Kb,
                __bf16* __restrict__ Vt) {
  const int tile = blockIdx.x;               // 128 * 48 tiles
  const int tM = tile / (N3 / 64);
  const int tN = tile % (N3 / 64);
  const int lane = threadIdx.x;

  v8f acc[2][4] = {};
  const __bf16* aB0 = xb + (size_t)(tM * 32) * EMB;
  const __bf16* aB1 = aB0 + (size_t)16 * EMB;
  const __bf16* bB  = Wt + (size_t)(tN * 64) * EMB;

  v16bf a0A, a1A, bfA[4], a0B, a1B, bfB[4];

  // prologue: set A <- k = 0
  a0A = load_frag(aB0, EMB, lane);
  a1A = load_frag(aB1, EMB, lane);
#pragma unroll
  for (int nb = 0; nb < 4; ++nb)
    bfA[nb] = load_frag(bB + (size_t)(nb * 16) * EMB, EMB, lane);

  for (int k0 = 0; k0 < EMB; k0 += 64) {
    // set B <- k0 + 32 (always in range: EMB % 64 == 0)
    const int kB = k0 + 32;
    a0B = load_frag(aB0 + kB, EMB, lane);
    a1B = load_frag(aB1 + kB, EMB, lane);
#pragma unroll
    for (int nb = 0; nb < 4; ++nb)
      bfB[nb] = load_frag(bB + (size_t)(nb * 16) * EMB + kB, EMB, lane);

    // compute with set A
#pragma unroll
    for (int nb = 0; nb < 4; ++nb) acc[0][nb] = wmma_bf16(a0A, bfA[nb], acc[0][nb]);
#pragma unroll
    for (int nb = 0; nb < 4; ++nb) acc[1][nb] = wmma_bf16(a1A, bfA[nb], acc[1][nb]);

    // set A <- k0 + 64 (dummy reload of k=0 on last iteration)
    const int kN = (k0 + 64 < EMB) ? k0 + 64 : 0;
    a0A = load_frag(aB0 + kN, EMB, lane);
    a1A = load_frag(aB1 + kN, EMB, lane);
#pragma unroll
    for (int nb = 0; nb < 4; ++nb)
      bfA[nb] = load_frag(bB + (size_t)(nb * 16) * EMB + kN, EMB, lane);

    // compute with set B
#pragma unroll
    for (int nb = 0; nb < 4; ++nb) acc[0][nb] = wmma_bf16(a0B, bfB[nb], acc[0][nb]);
#pragma unroll
    for (int nb = 0; nb < 4; ++nb) acc[1][nb] = wmma_bf16(a1B, bfB[nb], acc[1][nb]);
  }

  const int mofs = (lane >= 16) ? 8 : 0;
  const int n = lane & 15;
#pragma unroll
  for (int mb = 0; mb < 2; ++mb) {
#pragma unroll
    for (int nb = 0; nb < 4; ++nb) {
      int ch = tN * 64 + nb * 16 + n;        // 0..3071
      int part = ch >> 10;                   // 0=Q 1=K 2=V
      int cc = ch & 1023;
      int h = cc >> 6, d = cc & 63;
#pragma unroll
      for (int r = 0; r < 8; ++r) {
        int token = tM * 32 + mb * 16 + mofs + r;
        int bb = token >> 11, t = token & (T_SZ - 1);
        __bf16 v = (__bf16)acc[mb][nb][r];
        size_t bh = (size_t)(bb * H_CNT + h);
        if (part == 0)      Qb[(bh * T_SZ + t) * D_SZ + d] = v;
        else if (part == 1) Kb[(bh * T_SZ + t) * D_SZ + d] = v;
        else                Vt[(bh * D_SZ + d) * T_SZ + t] = v;
      }
    }
  }
}

// ===========================================================================
// Pass 1: per-column softmax stats, one wave per 16 columns of one (b,h).
// q-loop unrolled by 2 with ping-pong Q fragments; second tile guarded.
// ===========================================================================
__global__ void __launch_bounds__(32)
col_stats_kernel(const __bf16* __restrict__ Qb, const __bf16* __restrict__ Kb,
                 float* __restrict__ cmax, float* __restrict__ cinv) {
  const int id = blockIdx.x;                 // 32 * 128
  const int bh = id >> 7, kblk = id & 127, k0 = kblk * 16;
  const int lane = threadIdx.x;
  const __bf16* Qh = Qb + (size_t)bh * T_SZ * D_SZ;
  const __bf16* Kh = Kb + (size_t)bh * T_SZ * D_SZ;

  const v16bf bk0 = load_frag(Kh + (size_t)k0 * D_SZ, D_SZ, lane);
  const v16bf bk1 = load_frag(Kh + (size_t)k0 * D_SZ + 32, D_SZ, lane);

  const int n = lane & 15, mofs = (lane >= 16) ? 8 : 0;
  const int gk = k0 + n;
  float mrun = -__builtin_inff(), srun = 0.f;

  // online update for one 16-query tile (ok==false -> no contribution)
  auto step = [&](int q0, bool ok, v16bf a0t, v16bf a1t) {
    v8f c = {};
    c = wmma_bf16(a0t, bk0, c);
    c = wmma_bf16(a1t, bk1, c);
    float sv[8], tmax = -__builtin_inff();
#pragma unroll
    for (int r = 0; r < 8; ++r) {
      int gq = q0 + mofs + r;
      sv[r] = (ok && gk <= gq) ? c[r] * SCALE : -__builtin_inff();
      tmax = fmaxf(tmax, sv[r]);
    }
    tmax = fmaxf(tmax, xor16(tmax));
    float mnew = fmaxf(mrun, tmax);          // finite from first tile on
    float psum = 0.f;
#pragma unroll
    for (int r = 0; r < 8; ++r) psum += __expf(sv[r] - mnew);
    psum += xor16(psum);
    srun = srun * __expf(mrun - mnew) + psum;
    mrun = mnew;
  };

  // prologue: set A <- tile kblk (always valid)
  v16bf a0A = load_frag(Qh + (size_t)(kblk * 16) * D_SZ, D_SZ, lane);
  v16bf a1A = load_frag(Qh + (size_t)(kblk * 16) * D_SZ + 32, D_SZ, lane);

  for (int qt = kblk; qt < NQT; qt += 2) {
    const int q1 = qt + 1;
    const bool ok1 = q1 < NQT;
    const int q1c = ok1 ? q1 * 16 : 0;
    v16bf a0B = load_frag(Qh + (size_t)q1c * D_SZ, D_SZ, lane);
    v16bf a1B = load_frag(Qh + (size_t)q1c * D_SZ + 32, D_SZ, lane);

    step(qt * 16, true, a0A, a1A);

    const int q2 = qt + 2;
    const int q2c = (q2 < NQT) ? q2 * 16 : 0;
    a0A = load_frag(Qh + (size_t)q2c * D_SZ, D_SZ, lane);
    a1A = load_frag(Qh + (size_t)q2c * D_SZ + 32, D_SZ, lane);

    step(q1c, ok1, a0B, a1B);
  }

  if (lane < 16) {
    cmax[(size_t)bh * T_SZ + gk] = mrun;
    cinv[(size_t)bh * T_SZ + gk] = 1.0f / srun;
  }
}

// ===========================================================================
// Pass 2: one wave per 16-query tile; kb-loop unrolled by 2 (step 64) with
// ping-pong K fragment sets; V fragments loaded at sub-block top (their use
// is after the S-WMMA + exp + LDS-transpose phase).
// ===========================================================================
__global__ void __launch_bounds__(32)
attn_out_kernel(const __bf16* __restrict__ Qb, const __bf16* __restrict__ Kb,
                const __bf16* __restrict__ Vt, const float* __restrict__ cmax,
                const float* __restrict__ cinv, float* __restrict__ out) {
  const int id = blockIdx.x;                 // 32 * 128
  const int bh = id >> 7, qblk = id & 127, q0 = qblk * 16;
  const int lane = threadIdx.x;
  const __bf16* Qh  = Qb + (size_t)bh * T_SZ * D_SZ;
  const __bf16* Kh  = Kb + (size_t)bh * T_SZ * D_SZ;
  const __bf16* Vth = Vt + (size_t)bh * D_SZ * T_SZ;
  const float* cm = cmax + (size_t)bh * T_SZ;
  const float* ci = cinv + (size_t)bh * T_SZ;

  const v16bf a0 = load_frag(Qh + (size_t)q0 * D_SZ, D_SZ, lane);
  const v16bf a1 = load_frag(Qh + (size_t)q0 * D_SZ + 32, D_SZ, lane);

  __shared__ __bf16 pt[16 * 32];
  v8f o0 = {}, o1 = {}, o2 = {}, o3 = {};
  const int n = lane & 15, mofs = (lane >= 16) ? 8 : 0;
  const int lim = q0 + 16;                   // exclusive causal k bound

  auto loadK = [&](int kbc, v16bf (&kf)[4]) {
    kf[0] = load_frag(Kh + (size_t)kbc * D_SZ, D_SZ, lane);
    kf[1] = load_frag(Kh + (size_t)kbc * D_SZ + 32, D_SZ, lane);
    kf[2] = load_frag(Kh + (size_t)(kbc + 16) * D_SZ, D_SZ, lane);
    kf[3] = load_frag(Kh + (size_t)(kbc + 16) * D_SZ + 32, D_SZ, lane);
  };

  // one 32-column sub-block: S -> p -> LDS transpose -> O += P@V
  auto process = [&](int kbc, bool ok, const v16bf (&kf)[4]) {
    v16bf vf0 = load_frag(Vth + (size_t)0  * T_SZ + kbc, T_SZ, lane);
    v16bf vf1 = load_frag(Vth + (size_t)16 * T_SZ + kbc, T_SZ, lane);
    v16bf vf2 = load_frag(Vth + (size_t)32 * T_SZ + kbc, T_SZ, lane);
    v16bf vf3 = load_frag(Vth + (size_t)48 * T_SZ + kbc, T_SZ, lane);

    v8f c0 = {}, c1 = {};
    c0 = wmma_bf16(a0, kf[0], c0);
    c0 = wmma_bf16(a1, kf[1], c0);
    c1 = wmma_bf16(a0, kf[2], c1);
    c1 = wmma_bf16(a1, kf[3], c1);

    const int gk0 = kbc + n, gk1 = kbc + 16 + n;   // in-bounds (kbc clamped)
    const float m0 = cm[gk0], i0 = ci[gk0];
    const float m1 = cm[gk1], i1 = ci[gk1];
#pragma unroll
    for (int r = 0; r < 8; ++r) {
      int gq = q0 + mofs + r;
      int mrow = mofs + r;
      float p0 = (ok && gk0 <= gq) ? __expf(c0[r] * SCALE - m0) * i0 : 0.f;
      float p1 = (ok && gk1 <= gq) ? __expf(c1[r] * SCALE - m1) * i1 : 0.f;
      pt[mrow * 32 + n]      = (__bf16)p0;
      pt[mrow * 32 + 16 + n] = (__bf16)p1;
    }
    __syncthreads();                         // single-wave WG: cheap ordering
    v16bf pa;
    {
      const __bf16* p = &pt[(lane & 15) * 32 + ((lane < 16) ? 0 : 8)];
#pragma unroll
      for (int i = 0; i < 8; ++i) { pa[i] = p[i]; pa[i + 8] = p[i + 16]; }
    }
    __syncthreads();

    o0 = wmma_bf16(pa, vf0, o0);
    o1 = wmma_bf16(pa, vf1, o1);
    o2 = wmma_bf16(pa, vf2, o2);
    o3 = wmma_bf16(pa, vf3, o3);
  };

  v16bf kfA[4], kfB[4];
  loadK(0, kfA);                             // prologue

  for (int kb = 0; kb < lim; kb += 64) {
    const int kbB = kb + 32;
    const bool okB = kbB < lim;
    const int kbBc = okB ? kbB : 0;
    loadK(kbBc, kfB);

    process(kb, true, kfA);

    const int kbN = kb + 64;
    const int kbNc = (kbN < lim) ? kbN : 0;
    loadK(kbNc, kfA);

    process(kbBc, okB, kfB);
  }

  const int b = bh >> 4, h = bh & 15;
#pragma unroll
  for (int r = 0; r < 8; ++r) {
    int gq = q0 + mofs + r;
    size_t base = ((size_t)(b * T_SZ + gq)) * EMB + h * D_SZ + n;
    out[base + 0]  = o0[r];
    out[base + 16] = o1[r];
    out[base + 32] = o2[r];
    out[base + 48] = o3[r];
  }
}

// ===========================================================================
extern "C" void kernel_launch(void* const* d_in, const int* in_sizes, int n_in,
                              void* d_out, int out_size, void* d_ws, size_t ws_size,
                              hipStream_t stream) {
  const float* x  = (const float*)d_in[0];   // (2,2048,1024) f32
  const float* Ws = (const float*)d_in[1];   // (1024,3072)  f32
  float* out = (float*)d_out;                // (2,2048,1024) f32

  uint8_t* ws = (uint8_t*)d_ws;
  __bf16* xb = (__bf16*)(ws);                               // 8 MB
  __bf16* Wt = (__bf16*)(ws + (size_t)8  * 1024 * 1024);    // 6 MB
  __bf16* Qb = (__bf16*)(ws + (size_t)14 * 1024 * 1024);    // 8 MB
  __bf16* Kb = (__bf16*)(ws + (size_t)22 * 1024 * 1024);    // 8 MB
  __bf16* Vt = (__bf16*)(ws + (size_t)30 * 1024 * 1024);    // 8 MB (transposed)
  float* cmax = (float*)(ws + (size_t)38 * 1024 * 1024);    // 256 KB
  float* cinv = (float*)(ws + (size_t)38 * 1024 * 1024 + 256 * 1024);

  const int nx = M_TOK * EMB;
  cvt_x_kernel<<<(nx + 255) / 256, 256, 0, stream>>>(x, xb, nx);
  cvt_w_kernel<<<(N3 * EMB + 255) / 256, 256, 0, stream>>>(Ws, Wt);

  qkv_gemm_kernel<<<(M_TOK / 32) * (N3 / 64), 32, 0, stream>>>(xb, Wt, Qb, Kb, Vt);

  col_stats_kernel<<<B_SZ * H_CNT * (T_SZ / 16), 32, 0, stream>>>(Qb, Kb, cmax, cinv);

  attn_out_kernel<<<B_SZ * H_CNT * (T_SZ / 16), 32, 0, stream>>>(Qb, Kb, Vt, cmax,
                                                                 cinv, out);
}
